// Transformer_68058051772628
// MI455X (gfx1250) — matrix operand order repaired and checked by reading, not compiled, not run on hardware
//
#include <hip/hip_runtime.h>

#define BATCH   32
#define JTOK    17
#define P0      256
#define DMODEL  768
#define NHEAD   12
#define DHEAD   64
#define NDEPTH  12
#define MLPDIM  3072
#define SCALE_ATTN 0.03608439182435161f   // 768^-0.5
#define LN_EPS     1e-5f

typedef __attribute__((ext_vector_type(16))) __bf16 v16bf;
typedef __attribute__((ext_vector_type(8)))  __bf16 v8bf;
typedef __attribute__((ext_vector_type(8)))  float  v8f;
typedef __attribute__((ext_vector_type(4)))  float  f4;

// ---------------- helpers ----------------
__device__ __forceinline__ unsigned short f2bf_u(float f) {
    unsigned int u = __float_as_uint(f);
    unsigned int r = u + 0x7FFFu + ((u >> 16) & 1u);   // round-to-nearest-even
    return (unsigned short)(r >> 16);
}
__device__ __forceinline__ __bf16 f2bf(float f) {
    unsigned short u = f2bf_u(f);
    return __builtin_bit_cast(__bf16, u);
}
#define SHUF16(lo, hi) __builtin_shufflevector(lo, hi, 0,1,2,3,4,5,6,7,8,9,10,11,12,13,14,15)

__device__ __forceinline__ unsigned lds_off32(const void* p) {
    // LDS aperture: LDS offset is the low 32 bits of the address
    return (unsigned)(unsigned long long)p;
}

// ---------------- small utility kernels ----------------
__global__ void copyf_kernel(const float* __restrict__ src, float* __restrict__ dst, size_t count) {
    size_t i = (size_t)blockIdx.x * blockDim.x + threadIdx.x;
    size_t stride = (size_t)gridDim.x * blockDim.x;
    for (; i < count; i += stride) dst[i] = src[i];
}

__global__ void zerof_kernel(float* __restrict__ dst, size_t count) {
    size_t i = (size_t)blockIdx.x * blockDim.x + threadIdx.x;
    size_t stride = (size_t)gridDim.x * blockDim.x;
    for (; i < count; i += stride) dst[i] = 0.0f;
}

__global__ void conv_bf16_kernel(const float* __restrict__ src, unsigned short* __restrict__ dst, size_t count) {
    size_t i = (size_t)blockIdx.x * blockDim.x + threadIdx.x;
    size_t stride = (size_t)gridDim.x * blockDim.x;
    for (; i < count; i += stride) dst[i] = f2bf_u(src[i]);
}

__global__ void bcast_pos_kernel(const float* __restrict__ pos_in, float* __restrict__ pos_out) {
    int p = blockIdx.x, b = blockIdx.y;
    size_t d = ((size_t)(b * P0 + p)) * DMODEL;
    size_t s = (size_t)p * DMODEL;
    for (int c = threadIdx.x; c < DMODEL; c += 256) pos_out[d + c] = pos_in[s + c];
}

__global__ void add_pos_kernel(float* __restrict__ x, const float* __restrict__ pos, int n, int curP) {
    int p = blockIdx.x, b = blockIdx.y;
    size_t xo = ((size_t)(b * n + JTOK + p)) * DMODEL;
    size_t po = ((size_t)(b * curP + p)) * DMODEL;
    for (int c = threadIdx.x; c < DMODEL; c += 256) x[xo + c] += pos[po + c];
}

// ---------------- LayerNorm (f32 in -> bf16 out) ----------------
__global__ void __launch_bounds__(256) ln_kernel(const float* __restrict__ x,
                                                 const float* __restrict__ g,
                                                 const float* __restrict__ beta,
                                                 unsigned short* __restrict__ out) {
    __shared__ float red[256];
    size_t base = (size_t)blockIdx.x * DMODEL;
    int t = threadIdx.x;
    float v0 = x[base + t], v1 = x[base + t + 256], v2 = x[base + t + 512];
    red[t] = v0 + v1 + v2;
    __syncthreads();
    for (int o = 128; o > 0; o >>= 1) { if (t < o) red[t] += red[t + o]; __syncthreads(); }
    float mean = red[0] * (1.0f / DMODEL);
    __syncthreads();
    float d0 = v0 - mean, d1 = v1 - mean, d2 = v2 - mean;
    red[t] = d0 * d0 + d1 * d1 + d2 * d2;
    __syncthreads();
    for (int o = 128; o > 0; o >>= 1) { if (t < o) red[t] += red[t + o]; __syncthreads(); }
    float rstd = rsqrtf(red[0] * (1.0f / DMODEL) + LN_EPS);
    out[base + t]       = f2bf_u(d0 * rstd * g[t]       + beta[t]);
    out[base + t + 256] = f2bf_u(d1 * rstd * g[t + 256] + beta[t + 256]);
    out[base + t + 512] = f2bf_u(d2 * rstd * g[t + 512] + beta[t + 512]);
}

// ---------------- generic WMMA GEMM: C[M,N] = A[M,K]*W[K,N] (+bias)(+res)(gelu) ----------------
// block: 256 threads = 8 waves; tile: 64 (M) x 32 (N); each wave one 16x16 WMMA accumulator.
// A tile staged global->LDS with async DMA (global_load_async_to_lds_b128, ASYNCcnt).
// W tile staged transposed so both fragments are contiguous b128 LDS loads.
__global__ void __launch_bounds__(256) gemm_wmma_kernel(
    const unsigned short* __restrict__ A, const unsigned short* __restrict__ W,
    const float* __restrict__ bias, const float* __restrict__ residual,
    float* __restrict__ outF, unsigned short* __restrict__ outBF,
    int M, int N, int K, int gelu)
{
    __shared__ __align__(16) unsigned short As[64][40];   // row pitch 80B, payload 64B
    __shared__ __align__(16) unsigned short Wt[32][40];   // transposed [n][k]
    int tid = threadIdx.x;
    int lane = tid & 31, wid = tid >> 5;
    int wm = wid & 3, wn = wid >> 2;
    int half = lane >> 4, l15 = lane & 15;
    int bm = blockIdx.x, bn = blockIdx.y;

    // per-thread async A chunk: one b128 (8 bf16) per thread per K-step
    int ar = tid >> 2, ach = tid & 3;                     // 64 rows x 4 chunks
    int agm = bm * 64 + ar; if (agm > M - 1) agm = M - 1;
    const unsigned short* aGlob = A + (size_t)agm * K + ach * 8;
    unsigned aLds = lds_off32(&As[ar][ach * 8]);

    // per-thread W slice: 4 consecutive elements (one u64 load), stored transposed
    int wr = tid >> 3, wc4 = (tid & 7) * 4;               // 32 rows x 8 quads

    v8f acc = {};
    for (int k0 = 0; k0 < K; k0 += 32) {
        // --- A: async global -> LDS DMA (16B per thread) ---
        {
            unsigned long long ga = (unsigned long long)(aGlob + k0);
            asm volatile("global_load_async_to_lds_b128 %0, %1, off"
                         :: "v"(aLds), "v"(ga) : "memory");
        }
        // --- W: vector load + transposed LDS store ---
        {
            const unsigned short* wp = W + (size_t)(k0 + wr) * N + bn * 32 + wc4;
            unsigned long long wv = *(const unsigned long long*)wp;
            Wt[wc4 + 0][wr] = (unsigned short)(wv);
            Wt[wc4 + 1][wr] = (unsigned short)(wv >> 16);
            Wt[wc4 + 2][wr] = (unsigned short)(wv >> 32);
            Wt[wc4 + 3][wr] = (unsigned short)(wv >> 48);
        }
        if (k0 + 32 < K)
            __builtin_prefetch(W + (size_t)(k0 + 32 + wr) * N + bn * 32, 0, 2);

        asm volatile("s_wait_asynccnt 0x0" ::: "memory");
        __syncthreads();

        // fragments: A layout k=(e<8?e:e+8)+8*half -> two contiguous runs of 8
    //            B layout k=(e<8?e:e+8)+8*half -> two contiguous runs of 8
        v8bf alo = *(const v8bf*)&As[wm * 16 + l15][half * 8];
        v8bf ahi = *(const v8bf*)&As[wm * 16 + l15][16 + half * 8];
        v8bf blo = *(const v8bf*)&Wt[wn * 16 + l15][half * 8];
        v8bf bhi = *(const v8bf*)&Wt[wn * 16 + l15][16 + half * 8];
        v16bf af  = SHUF16(alo, ahi);
        v16bf bfv = SHUF16(blo, bhi);
        acc = __builtin_amdgcn_wmma_f32_16x16x32_bf16(false, af, false, bfv, (short)0, acc, false, false);
        __syncthreads();
    }

    int nG = bn * 32 + wn * 16 + l15;
#pragma unroll
    for (int r = 0; r < 8; r++) {
        int mG = bm * 64 + wm * 16 + half * 8 + r;        // C/D: m = r + 8*half
        if (mG < M) {
            float v = acc[r];
            if (bias)     v += bias[nG];
            if (residual) v += residual[(size_t)mG * N + nG];
            if (gelu)     v = 0.5f * v * (1.0f + erff(v * 0.7071067811865475f));
            if (outF)  outF[(size_t)mG * N + nG] = v;
            if (outBF) outBF[(size_t)mG * N + nG] = f2bf_u(v);
        }
    }
}

// ---------------- fused attention: one wave per (b, h, 16-query tile) ----------------
// qkv layout: [b][token][3*D], q|k|v concatenated; head h uses cols h*64 .. h*64+63.
__global__ void __launch_bounds__(32) attn_kernel(const float* __restrict__ qkv,
                                                  unsigned short* __restrict__ obf,
                                                  float* __restrict__ human,
                                                  int n, int doHuman)
{
    __shared__ __align__(16) float sS[16][288];
    int tile = blockIdx.x, h = blockIdx.y, b = blockIdx.z;
    int lane = threadIdx.x, half = lane >> 4, l15 = lane & 15;
    int npad = (n + 31) & ~31;
    int nt16 = (n + 15) >> 4;

    for (int t = lane; t < 16 * 288; t += 32) (&sS[0][0])[t] = -1e30f;
    __syncthreads();

    const size_t row = 3 * (size_t)DMODEL;
    const float* Q  = qkv + (size_t)b * n * row + (size_t)h * DHEAD;
    const float* Kp = Q + DMODEL;
    const float* Vp = Q + 2 * DMODEL;

    // Q fragments (A layout: two contiguous runs of 8 floats per 32-wide K step)
    int qt = tile * 16 + l15; if (qt >= n) qt = n - 1;
    v16bf a0, a1;
    {
        const float* qr = Q + (size_t)qt * row;
        f4 t0 = *(const f4*)(qr + half * 8);
        f4 t1 = *(const f4*)(qr + half * 8 + 4);
        f4 t2 = *(const f4*)(qr + 16 + half * 8);
        f4 t3 = *(const f4*)(qr + 16 + half * 8 + 4);
        f4 u0 = *(const f4*)(qr + 32 + half * 8);
        f4 u1 = *(const f4*)(qr + 32 + half * 8 + 4);
        f4 u2 = *(const f4*)(qr + 48 + half * 8);
        f4 u3 = *(const f4*)(qr + 48 + half * 8 + 4);
#pragma unroll
        for (int e = 0; e < 16; e++) {
            float qv = (e < 4) ? t0[e] : (e < 8) ? t1[e - 4] : (e < 12) ? t2[e - 8] : t3[e - 12];
            float qw = (e < 4) ? u0[e] : (e < 8) ? u1[e - 4] : (e < 12) ? u2[e - 8] : u3[e - 12];
            a0[e] = f2bf(qv);
            a1[e] = f2bf(qw);
        }
    }

    // scores S = Q K^T * scale   (B layout: two contiguous runs of 8 floats, same as A)
    for (int c = 0; c < nt16; c++) {
        int kt = c * 16 + l15; if (kt >= n) kt = n - 1;
        const float* kr = Kp + (size_t)kt * row;
        f4 k0a = *(const f4*)(kr + half * 8);
        f4 k0b = *(const f4*)(kr + half * 8 + 4);
        f4 k0c = *(const f4*)(kr + 16 + half * 8);
        f4 k0d = *(const f4*)(kr + 16 + half * 8 + 4);
        f4 k1a = *(const f4*)(kr + 32 + half * 8);
        f4 k1b = *(const f4*)(kr + 32 + half * 8 + 4);
        f4 k1c = *(const f4*)(kr + 48 + half * 8);
        f4 k1d = *(const f4*)(kr + 48 + half * 8 + 4);
        v16bf b0, b1;
#pragma unroll
        for (int e = 0; e < 16; e++) {
            float kv = (e < 4) ? k0a[e] : (e < 8) ? k0b[e - 4] : (e < 12) ? k0c[e - 8] : k0d[e - 12];
            float kw = (e < 4) ? k1a[e] : (e < 8) ? k1b[e - 4] : (e < 12) ? k1c[e - 8] : k1d[e - 12];
            b0[e] = f2bf(kv);
            b1[e] = f2bf(kw);
        }
        v8f s = {};
        s = __builtin_amdgcn_wmma_f32_16x16x32_bf16(false, a0, false, b0, (short)0, s, false, false);
        s = __builtin_amdgcn_wmma_f32_16x16x32_bf16(false, a1, false, b1, (short)0, s, false, false);
        int col = c * 16 + l15;
#pragma unroll
        for (int r = 0; r < 8; r++)
            if (col < n) sS[half * 8 + r][col] = s[r] * SCALE_ATTN;
    }
    __syncthreads();

    // row softmax (padded cols hold -1e30 -> exp = 0)
    for (int r = 0; r < 16; r++) {
        float mx = -1e30f;
        for (int c2 = lane; c2 < npad; c2 += 32) mx = fmaxf(mx, sS[r][c2]);
        for (int o = 16; o > 0; o >>= 1) mx = fmaxf(mx, __shfl_xor(mx, o, 32));
        float sum = 0.0f;
        for (int c2 = lane; c2 < npad; c2 += 32) { float e = __expf(sS[r][c2] - mx); sS[r][c2] = e; sum += e; }
        for (int o = 16; o > 0; o >>= 1) sum += __shfl_xor(sum, o, 32);
        float inv = 1.0f / sum;
        for (int c2 = lane; c2 < npad; c2 += 32) sS[r][c2] *= inv;
    }
    __syncthreads();

    // token importance: human[b,p] += (1/H) * sum_{j<J} attn[j, J+p]
    if (doHuman) {
        for (int r = 0; r < 16; r++) {
            int t = tile * 16 + r;
            if (t < JTOK)
                for (int c2 = JTOK + lane; c2 < n; c2 += 32)
                    atomicAdd(&human[b * P0 + (c2 - JTOK)], sS[r][c2] * (1.0f / NHEAD));
        }
    }

    // O = P * V  (K-dim = tokens, steps of 32; 4 accumulators cover d=64)
    v8f acc[4] = {};
    for (int kc = 0; kc < npad; kc += 32) {
        const float* pr = &sS[l15][kc];
        f4 p0 = *(const f4*)(pr + half * 8);
        f4 p1 = *(const f4*)(pr + half * 8 + 4);
        f4 p2 = *(const f4*)(pr + 16 + half * 8);
        f4 p3 = *(const f4*)(pr + 16 + half * 8 + 4);
        v16bf ap;
#pragma unroll
        for (int e = 0; e < 16; e++) {
            float pv = (e < 4) ? p0[e] : (e < 8) ? p1[e - 4] : (e < 12) ? p2[e - 8] : p3[e - 12];
            ap[e] = f2bf(pv);
        }
#pragma unroll
        for (int dn = 0; dn < 4; dn++) {
            v16bf bv;
#pragma unroll
            for (int e = 0; e < 16; e++) {
                int kb = ((e < 8) ? e : e + 8) + half * 8;
                int vt = kc + kb; if (vt >= n) vt = n - 1;   // P==0 beyond n, value irrelevant
                bv[e] = f2bf(Vp[(size_t)vt * row + dn * 16 + l15]);
            }
            acc[dn] = __builtin_amdgcn_wmma_f32_16x16x32_bf16(false, ap, false, bv, (short)0, acc[dn], false, false);
        }
    }
#pragma unroll
    for (int dn = 0; dn < 4; dn++)
#pragma unroll
        for (int r = 0; r < 8; r++) {
            int t = tile * 16 + half * 8 + r;
            if (t < n)
                obf[((size_t)(b * n + t)) * DMODEL + h * DHEAD + dn * 16 + l15] = f2bf_u(acc[dn][r]);
        }
}

// ---------------- stable descending top-k via rank counting ----------------
__global__ void __launch_bounds__(256) topk_kernel(const float* __restrict__ human,
                                                   int* __restrict__ idx, int curP, int keep) {
    __shared__ float sc[256];
    int b = blockIdx.x, i = threadIdx.x;
    sc[i] = (i < curP) ? human[b * P0 + i] : -1e30f;
    __syncthreads();
    if (i < curP) {
        float si = sc[i];
        int rank = 0;
        for (int j = 0; j < curP; j++) {
            float sj = sc[j];
            rank += (sj > si) || (sj == si && j < i);
        }
        if (rank < keep) idx[b * P0 + rank] = i;
    }
}

// ---------------- gather kept tokens (x) and their positions ----------------
__global__ void gather_kernel(const float* __restrict__ xin, float* __restrict__ xout,
                              const float* __restrict__ posin, float* __restrict__ posout,
                              const int* __restrict__ idx, int n_old, int n_new, int p_old) {
    int t = blockIdx.x, b = blockIdx.y;
    int src = t;
    if (t >= JTOK) src = JTOK + idx[b * P0 + (t - JTOK)];
    size_t so = ((size_t)(b * n_old + src)) * DMODEL;
    size_t dof = ((size_t)(b * n_new + t)) * DMODEL;
    for (int c = threadIdx.x; c < DMODEL; c += 256) xout[dof + c] = xin[so + c];
    if (t >= JTOK) {
        int p = idx[b * P0 + (t - JTOK)];
        size_t ps = ((size_t)(b * p_old + p)) * DMODEL;
        size_t pd = ((size_t)(b * (n_new - JTOK) + (t - JTOK))) * DMODEL;
        for (int c = threadIdx.x; c < DMODEL; c += 256) posout[pd + c] = posin[ps + c];
    }
}

// ---------------- host orchestration ----------------
extern "C" void kernel_launch(void* const* d_in, const int* in_sizes, int n_in,
                              void* d_out, int out_size, void* d_ws, size_t ws_size,
                              hipStream_t stream) {
    (void)in_sizes; (void)n_in; (void)ws_size;
    const float* xin  = (const float*)d_in[0];
    const float* posin= (const float*)d_in[1];
    const float* ln1g = (const float*)d_in[2];
    const float* ln1b = (const float*)d_in[3];
    const float* qkvw = (const float*)d_in[4];
    const float* outw = (const float*)d_in[5];
    const float* outb = (const float*)d_in[6];
    const float* ln2g = (const float*)d_in[7];
    const float* ln2b = (const float*)d_in[8];
    const float* ffw1 = (const float*)d_in[9];
    const float* ffb1 = (const float*)d_in[10];
    const float* ffw2 = (const float*)d_in[11];
    const float* ffb2 = (const float*)d_in[12];

    const size_t MAXN = JTOK + P0;                 // 273
    const size_t MAXTOK = (size_t)BATCH * MAXN;

    size_t off = 0;
    char* base = (char*)d_ws;
    auto alloc = [&](size_t bytes) -> void* {
        void* p = (void*)(base + off);
        off += (bytes + 255) & ~(size_t)255;
        return p;
    };
    float* x_a   = (float*)alloc(MAXTOK * DMODEL * 4);
    float* x_b   = (float*)alloc(MAXTOK * DMODEL * 4);
    float* pos_a = (float*)alloc((size_t)BATCH * P0 * DMODEL * 4);
    float* pos_b = (float*)alloc((size_t)BATCH * P0 * DMODEL * 4);
    unsigned short* a_bf   = (unsigned short*)alloc(MAXTOK * DMODEL * 2);
    unsigned short* mlp_bf = (unsigned short*)alloc(MAXTOK * MLPDIM * 2);
    float* qkvbuf = (float*)alloc(MAXTOK * 3 * DMODEL * 4);
    unsigned short* w_qkv = (unsigned short*)alloc((size_t)DMODEL * 3 * DMODEL * 2);
    unsigned short* w_out = (unsigned short*)alloc((size_t)DMODEL * DMODEL * 2);
    unsigned short* w_ff1 = (unsigned short*)alloc((size_t)DMODEL * MLPDIM * 2);
    unsigned short* w_ff2 = (unsigned short*)alloc((size_t)MLPDIM * DMODEL * 2);
    float* human = (float*)alloc((size_t)BATCH * P0 * 4);
    int*   idxb  = (int*)alloc((size_t)BATCH * P0 * 4);

    float* xA = x_a; float* xB = x_b;
    float* pA = pos_a; float* pB = pos_b;

    copyf_kernel<<<2048, 256, 0, stream>>>(xin, xA, MAXTOK * DMODEL);
    bcast_pos_kernel<<<dim3(P0, BATCH), 256, 0, stream>>>(posin, pA);

    int curP = P0;
    for (int i = 0; i < NDEPTH; i++) {
        int n = JTOK + curP;
        size_t ntok = (size_t)BATCH * n;
        if (i > 0)
            add_pos_kernel<<<dim3(curP, BATCH), 256, 0, stream>>>(xA, pA, n, curP);

        conv_bf16_kernel<<<1024, 256, 0, stream>>>(qkvw + (size_t)i * DMODEL * 3 * DMODEL, w_qkv, (size_t)DMODEL * 3 * DMODEL);
        conv_bf16_kernel<<<512,  256, 0, stream>>>(outw + (size_t)i * DMODEL * DMODEL,     w_out, (size_t)DMODEL * DMODEL);
        conv_bf16_kernel<<<1024, 256, 0, stream>>>(ffw1 + (size_t)i * DMODEL * MLPDIM,     w_ff1, (size_t)DMODEL * MLPDIM);
        conv_bf16_kernel<<<1024, 256, 0, stream>>>(ffw2 + (size_t)i * MLPDIM * DMODEL,     w_ff2, (size_t)MLPDIM * DMODEL);

        // --- attention block ---
        ln_kernel<<<(unsigned)ntok, 256, 0, stream>>>(xA, ln1g + (size_t)i * DMODEL, ln1b + (size_t)i * DMODEL, a_bf);
        gemm_wmma_kernel<<<dim3((unsigned)((ntok + 63) / 64), (3 * DMODEL) / 32), 256, 0, stream>>>(
            a_bf, w_qkv, nullptr, nullptr, qkvbuf, nullptr, (int)ntok, 3 * DMODEL, DMODEL, 0);

        int isPrune = (i == 3 || i == 6 || i == 9) ? 1 : 0;
        if (isPrune) zerof_kernel<<<32, 256, 0, stream>>>(human, (size_t)BATCH * P0);

        attn_kernel<<<dim3((unsigned)((n + 15) / 16), NHEAD, BATCH), 32, 0, stream>>>(
            qkvbuf, a_bf, human, n, isPrune);

        // out projection fused with residual: x = attn @ out_w + out_b + x
        gemm_wmma_kernel<<<dim3((unsigned)((ntok + 63) / 64), DMODEL / 32), 256, 0, stream>>>(
            a_bf, w_out, outb + (size_t)i * DMODEL, xA, xA, nullptr, (int)ntok, DMODEL, DMODEL, 0);

        // --- prune ---
        if (isPrune) {
            int keep = (7 * curP + 9) / 10;      // ceil(0.7 * curP)
            int n_new = JTOK + keep;
            topk_kernel<<<BATCH, 256, 0, stream>>>(human, idxb, curP, keep);
            gather_kernel<<<dim3((unsigned)n_new, BATCH), 256, 0, stream>>>(
                xA, xB, pA, pB, idxb, n, n_new, curP);
            float* t1 = xA; xA = xB; xB = t1;
            float* t2 = pA; pA = pB; pB = t2;
            curP = keep; n = n_new; ntok = (size_t)BATCH * n;
        }

        // --- MLP block ---
        ln_kernel<<<(unsigned)ntok, 256, 0, stream>>>(xA, ln2g + (size_t)i * DMODEL, ln2b + (size_t)i * DMODEL, a_bf);
        gemm_wmma_kernel<<<dim3((unsigned)((ntok + 63) / 64), MLPDIM / 32), 256, 0, stream>>>(
            a_bf, w_ff1, ffb1 + (size_t)i * MLPDIM, nullptr, nullptr, mlp_bf, (int)ntok, MLPDIM, DMODEL, 1);
        gemm_wmma_kernel<<<dim3((unsigned)((ntok + 63) / 64), DMODEL / 32), 256, 0, stream>>>(
            mlp_bf, w_ff2, ffb2 + (size_t)i * DMODEL, xA, xA, nullptr, (int)ntok, DMODEL, MLPDIM, 0);
    }

    copyf_kernel<<<2048, 256, 0, stream>>>(xA, (float*)d_out, (size_t)out_size);
}
